// QuantizedLinear_16123307229809
// MI455X (gfx1250) — compile-verified, matrix operand order checked by reading, not run on hardware
//
#include <hip/hip_runtime.h>

// ---------------------------------------------------------------------------
// QuantizedLinear forward for MI455X (gfx1250, wave32, WMMA).
//
//   y[m,n] = sum_k x[m,k] * W[n,k] + bias[n]
//   W[n, b*8+e] = centroids[assignments[b*OUT_F + n]][e]
//
// M=1024, N=4096, K=4096.  ~34.4 GFLOP, ~40 MB mandatory HBM traffic
// (x 16MB + assignments 8MB + out 16MB) -> memory floor ~1.7us @ 23.3 TB/s.
// bf16 hi/lo split inputs + 3x v_wmma_f32_16x16x32_bf16 per K=32 chunk give
// near-fp32 accuracy at ~2.7x the f32-WMMA rate, keeping us at the roofline.
// Staging uses v_perm_b32 pair-packing to keep the VALU stream (which
// co-executes with the XDL WMMA pipe) from becoming the bottleneck.
// ---------------------------------------------------------------------------

#define IN_F        4096
#define OUT_F       4096
#define N_TOKENS    1024
#define QBLOCK      8

#define TILE_M      128
#define TILE_N      128
#define TILE_K      32
#define LDS_STRIDE  20   // uints per row: 32 halves + pad -> 80B rows

typedef __attribute__((ext_vector_type(16))) __bf16        v16bf;
typedef __attribute__((ext_vector_type(8)))  float         v8f;
typedef __attribute__((ext_vector_type(4)))  unsigned int  uint4v;

union FragU { v16bf v; uint4v q[2]; };
union F16U  { float4 v4[4]; float f[16]; unsigned int u[16]; };
union F8U   { float4 v4[2]; float f[8];  unsigned int u[8];  };
union FB    { float f; unsigned int u; };

// Split a pair of f32 into packed-bf16 hi and lo words (even value -> [15:0]).
// hi = truncate-to-bf16 (top 16 bits), lo = bf16(f - hi): hi+lo captures the
// top 16 mantissa bits exactly. 6 VALU ops per 2 values, no sub-dword moves.
__device__ __forceinline__ void bf16_split_pair(float f0, float f1,
                                                unsigned int& hi01,
                                                unsigned int& lo01) {
    FB a; a.f = f0;
    FB b; b.f = f1;
    FB ha; ha.u = a.u & 0xFFFF0000u;
    FB hb; hb.u = b.u & 0xFFFF0000u;
    FB la; la.f = f0 - ha.f;
    FB lb; lb.f = f1 - hb.f;
    hi01 = __builtin_amdgcn_perm(b.u,  a.u,  0x07060302u);  // {hi16(f1),hi16(f0)}
    lo01 = __builtin_amdgcn_perm(lb.u, la.u, 0x07060302u);
}

// Load one 16x16(x32) bf16 operand fragment from an LDS row per the CDNA5
// 16-bit A/B layout: halves 0..7 = K[khalf..khalf+7], halves 8..15 =
// K[khalf+16..khalf+23], khalf = (lane>=16)?8:0. Rows store one packed K-pair
// per dword (even K low), so this is two ds_load_b128 at dword offsets.
__device__ __forceinline__ v16bf load_frag(const unsigned int* row, int kw) {
    FragU f;
    f.q[0] = *(const uint4v*)(row + kw);      // kw = khalf/2 : 0 or 4
    f.q[1] = *(const uint4v*)(row + kw + 8);  // +16 halves = +8 dwords
    return f.v;
}

__global__ __launch_bounds__(256)
void qlinear_bf16x3_wmma(const float* __restrict__ x,
                         const float* __restrict__ centroids,
                         const float* __restrict__ bias,
                         const int*   __restrict__ assignments,
                         float*       __restrict__ out)
{
    const int n_base = blockIdx.x * TILE_N;
    const int m_base = blockIdx.y * TILE_M;
    const int tid    = threadIdx.x;
    const int lane   = tid & 31;
    const int wave   = tid >> 5;

    // 8 waves arranged 4(M) x 2(N): each wave owns a 32x64 output sub-tile
    const int wm = wave & 3;   // M offset = wm*32
    const int wn = wave >> 2;  // N offset = wn*64

    const int lrow = lane & 15;
    const int kw   = (lane >> 4) << 2;   // dword K-offset: 0 or 4

    __shared__ __align__(16) unsigned int xs_hi[TILE_M][LDS_STRIDE];
    __shared__ __align__(16) unsigned int xs_lo[TILE_M][LDS_STRIDE];
    __shared__ __align__(16) unsigned int ws_hi[TILE_N][LDS_STRIDE];
    __shared__ __align__(16) unsigned int ws_lo[TILE_N][LDS_STRIDE];

    v8f acc[2][4] = {};   // [mi][ni] fp32 accumulators

    // staging roles (fixed per thread)
    const int xr  = tid >> 1;            // x row 0..127
    const int xcw = (tid & 1) << 3;      // dword col: 0 or 8 (16 floats each)
    const int wo  = tid & 127;           // weight output column 0..127
    const int wj  = (tid >> 7) << 1;     // first of two 8-wide K sub-blocks

    for (int kc = 0; kc < IN_F; kc += TILE_K) {
        // ---- stage x tile: f32 -> packed bf16 hi/lo -----------------------
        {
            const float* src = x + (size_t)(m_base + xr) * IN_F + kc + (xcw << 1);
            F16U u;
            const float4* s4 = (const float4*)src;
            u.v4[0] = s4[0]; u.v4[1] = s4[1]; u.v4[2] = s4[2]; u.v4[3] = s4[3];
            uint4v hq[2], lq[2];
            #pragma unroll
            for (int p = 0; p < 8; ++p) {
                unsigned int h, l;
                bf16_split_pair(u.f[2 * p], u.f[2 * p + 1], h, l);
                hq[p >> 2][p & 3] = h;
                lq[p >> 2][p & 3] = l;
            }
            *(uint4v*)&xs_hi[xr][xcw]     = hq[0];
            *(uint4v*)&xs_hi[xr][xcw + 4] = hq[1];
            *(uint4v*)&xs_lo[xr][xcw]     = lq[0];
            *(uint4v*)&xs_lo[xr][xcw + 4] = lq[1];
            // prefetch next chunk's x rows into cache (global_prefetch_b8)
            if (kc + TILE_K < IN_F)
                __builtin_prefetch(src + TILE_K, 0, 3);
        }
        // ---- decode weight tile: gather centroids, packed bf16 hi/lo ------
        {
            #pragma unroll
            for (int j2 = 0; j2 < 2; ++j2) {
                const int j   = wj + j2;                 // 0..3 (8 K-values each)
                const int blk = (kc >> 3) + j;           // global 8-wide block idx
                const int idx = assignments[(size_t)blk * OUT_F + n_base + wo];
                F8U c;
                const float4* cp = (const float4*)(centroids + (size_t)idx * QBLOCK);
                c.v4[0] = cp[0]; c.v4[1] = cp[1];
                uint4v hq, lq;
                #pragma unroll
                for (int p = 0; p < 4; ++p) {
                    unsigned int h, l;
                    bf16_split_pair(c.f[2 * p], c.f[2 * p + 1], h, l);
                    hq[p] = h;
                    lq[p] = l;
                }
                *(uint4v*)&ws_hi[wo][j << 2] = hq;
                *(uint4v*)&ws_lo[wo][j << 2] = lq;
            }
            if (kc + TILE_K < IN_F)
                __builtin_prefetch(assignments + (size_t)((kc + TILE_K) >> 3) * OUT_F
                                               + n_base + wo, 0, 3);
        }
        __syncthreads();

        // ---- compute: 8 tiles x 3 precision terms = 24 WMMAs --------------
        v16bf a_hi[2], a_lo[2], b_hi[4], b_lo[4];
        #pragma unroll
        for (int mi = 0; mi < 2; ++mi) {
            const int r = wm * 32 + mi * 16 + lrow;
            a_hi[mi] = load_frag(&xs_hi[r][0], kw);
            a_lo[mi] = load_frag(&xs_lo[r][0], kw);
        }
        #pragma unroll
        for (int ni = 0; ni < 4; ++ni) {
            const int r = wn * 64 + ni * 16 + lrow;
            b_hi[ni] = load_frag(&ws_hi[r][0], kw);
            b_lo[ni] = load_frag(&ws_lo[r][0], kw);
        }
        #pragma unroll
        for (int mi = 0; mi < 2; ++mi) {
            #pragma unroll
            for (int ni = 0; ni < 4; ++ni) {
                v8f c = acc[mi][ni];
                c = __builtin_amdgcn_wmma_f32_16x16x32_bf16(
                        false, a_hi[mi], false, b_hi[ni], (short)0, c, false, false);
                c = __builtin_amdgcn_wmma_f32_16x16x32_bf16(
                        false, a_hi[mi], false, b_lo[ni], (short)0, c, false, false);
                c = __builtin_amdgcn_wmma_f32_16x16x32_bf16(
                        false, a_lo[mi], false, b_hi[ni], (short)0, c, false, false);
                acc[mi][ni] = c;
            }
        }
        __syncthreads();   // protect LDS before next stage overwrites it
    }

    // ---- epilogue: bias add + store (C/D layout: elem i -> M = i + (lane>=16?8:0),
    //                                                N = lane & 15) ----------
    const int mofs = (lane >> 4) << 3;
    #pragma unroll
    for (int ni = 0; ni < 4; ++ni) {
        const int n  = n_base + wn * 64 + ni * 16 + (lane & 15);
        const float bv = bias[n];
        #pragma unroll
        for (int mi = 0; mi < 2; ++mi) {
            const int mrow = m_base + wm * 32 + mi * 16 + mofs;
            #pragma unroll
            for (int i = 0; i < 8; ++i) {
                out[(size_t)(mrow + i) * OUT_F + n] = acc[mi][ni][i] + bv;
            }
        }
    }
}

extern "C" void kernel_launch(void* const* d_in, const int* in_sizes, int n_in,
                              void* d_out, int out_size, void* d_ws, size_t ws_size,
                              hipStream_t stream) {
    // setup_inputs() order: x, centroids, bias, assignments, counts(unused)
    const float* x           = (const float*)d_in[0];
    const float* centroids   = (const float*)d_in[1];
    const float* bias        = (const float*)d_in[2];
    const int*   assignments = (const int*)d_in[3];
    float*       out         = (float*)d_out;

    dim3 grid(OUT_F / TILE_N, N_TOKENS / TILE_M);   // 32 x 8 = 256 workgroups
    dim3 block(256);                                 // 8 waves (wave32)
    qlinear_bf16x3_wmma<<<grid, block, 0, stream>>>(x, centroids, bias,
                                                    assignments, out);
}
